// MyRNN_33011118637831
// MI455X (gfx1250) — matrix operand order repaired and checked by reading, not compiled, not run on hardware
//
#include <hip/hip_runtime.h>

// Problem dims (from reference)
#define BATCH  4096
#define SEQ    80
#define EMBD   100
#define UNITS  64
#define NGATE  256          // 4*UNITS
#define K0TOT  192          // 128 (x padded) + 64 (h0)   layer0: [x || h0] @ [W0;U0]
#define K1TOT  128          // 64 (h0) + 64 (h1)          layer1: [h0 || h1] @ [W1;U1]
#define BM     16           // batch rows per workgroup (WMMA M)
#define LDA    384          // A-buffer row: x[0..128) | h0A[128..192) | h1A[192..256) | h0B[256..320) | h1B[320..384)
#define NTHREADS 128        // 4 waves (wave32); wave w owns units [16w,16w+16) x 4 gates

typedef __attribute__((ext_vector_type(16))) _Float16 v16h;
typedef __attribute__((ext_vector_type(8)))  _Float16 v8h;
typedef __attribute__((ext_vector_type(8)))  float    v8f;

// ---- LDS layout (dynamic shared memory, byte offsets) ----
constexpr size_t OFF_WT0 = 0;                                    // f16 [256][192] = 96KB (cols permuted)
constexpr size_t OFF_WT1 = OFF_WT0 + (size_t)NGATE * K0TOT * 2;  // f16 [256][128] = 64KB (cols permuted)
constexpr size_t OFF_AB  = OFF_WT1 + (size_t)NGATE * K1TOT * 2;  // f16 [16][384]  = 12KB
constexpr size_t SMEM_BYTES = OFF_AB + (size_t)BM * LDA * 2;     // 176,128 B < 320KB/WGP

__device__ __forceinline__ float sigmoidf_(float x) {
    return 1.0f / (1.0f + __expf(-x));
}

// A fragment: 16x32 f16 tile from row-major [M][LDA] LDS buffer.
// lanes 0-15 (M=lane) hold K = {k0..k0+7, k0+16..k0+23},
// lanes 16-31 (M=lane-16) hold K = {k0+8..k0+15, k0+24..k0+31}.
__device__ __forceinline__ v16h load_a_frag(const _Float16* A, int k0, int lane) {
    const int m  = lane & 15;
    const int ko = k0 + ((lane & 16) ? 8 : 0);
    const v8h* p = (const v8h*)(A + m * LDA + ko);
    v8h lo = p[0];          // 16B -> ds_load_b128
    v8h hi = p[2];
    v16h r;
#pragma unroll
    for (int i = 0; i < 8; ++i) { r[i] = lo[i]; r[8 + i] = hi[i]; }
    return r;
}

// B fragment: 32x16 f16 tile; weights stored transposed [N'][ldb] (K contiguous).
// lanes 0-15 (N=lane) hold K = {k0..k0+7, k0+16..k0+23}, lanes 16-31 hold K = {k0+8..k0+15, k0+24..k0+31}.
__device__ __forceinline__ v16h load_b_frag(const _Float16* Bt, int ldb, int n0, int k0, int lane) {
    const int n  = n0 + (lane & 15);
    const int ko = k0 + ((lane & 16) ? 8 : 0);
    const v8h* p = (const v8h*)(Bt + n * ldb + ko);
    v8h lo = p[0];
    v8h hi = p[2];
    v16h r;
#pragma unroll
    for (int i = 0; i < 8; ++i) { r[i] = lo[i]; r[8 + i] = hi[i]; }
    return r;
}

// One wave: 4 gate-tiles (i,f,g,o) for its 16 units; A-frag reused across the 4 WMMAs.
// abase[kk] gives the A-buffer column base of K-chunk kk (handles parity regions).
template <int NK>
__device__ __forceinline__ void gemm_gates(const _Float16* __restrict__ A,
                                           const int* abase,
                                           const _Float16* __restrict__ Bt, int ldb,
                                           int n0, int lane, v8f acc[4]) {
#pragma unroll
    for (int kk = 0; kk < NK; ++kk) {
        v16h a = load_a_frag(A, abase[kk], lane);
#pragma unroll
        for (int g = 0; g < 4; ++g) {
            v16h b = load_b_frag(Bt, ldb, n0 + g * 16, kk * 32, lane);
            acc[g] = __builtin_amdgcn_wmma_f32_16x16x32_f16(
                false, a, false, b, (short)0, acc[g], false, false);
        }
    }
}

// In-register LSTM cell update for 8 rows x 1 unit per lane; stores h (f16) to LDS column.
__device__ __forceinline__ void cell_update(const v8f acc[4], v8f& c, const float4 bb,
                                            _Float16* hcol /* &ab[mb*LDA + hwb + j] */) {
#pragma unroll
    for (int r = 0; r < 8; ++r) {
        const float zi = acc[0][r] + bb.x;
        const float zf = acc[1][r] + bb.y;
        const float zg = acc[2][r] + bb.z;
        const float zo = acc[3][r] + bb.w;
        const float cn = sigmoidf_(zf) * c[r] + sigmoidf_(zi) * tanhf(zg);
        const float hn = sigmoidf_(zo) * tanhf(cn);
        c[r] = cn;
        hcol[r * LDA] = (_Float16)hn;
    }
}

__global__ void __launch_bounds__(NTHREADS)
lstm2_fused(const int*   __restrict__ inputs,   // [B][T]
            const float* __restrict__ emb,      // [VOCAB][100]
            const float* __restrict__ W0,       // [100][256]
            const float* __restrict__ U0,       // [64][256]
            const float* __restrict__ b0,       // [256]
            const float* __restrict__ W1,       // [64][256]
            const float* __restrict__ U1,       // [64][256]
            const float* __restrict__ b1,       // [256]
            const float* __restrict__ Wout,     // [64]
            const float* __restrict__ bout,     // [1]
            float*       __restrict__ out)      // [B]
{
    extern __shared__ char smem[];
    _Float16* wt0 = (_Float16*)(smem + OFF_WT0);
    _Float16* wt1 = (_Float16*)(smem + OFF_WT1);
    _Float16* ab  = (_Float16*)(smem + OFF_AB);

    const int tid   = threadIdx.x;
    const int lane  = tid & 31;
    const int wave  = tid >> 5;            // 0..3
    const int lidx  = lane & 15;
    const int mb    = (lane & 16) ? 8 : 0; // row group of C/D layout
    const int j     = wave * 16 + lidx;    // unit owned by this lane
    const int n0    = wave * 64;           // permuted-column base of this wave's 4 tiles
    const int bbase = blockIdx.x * BM;
    const int mrow  = tid >> 3;            // 0..15 (gather row)
    const int ecol  = tid & 7;             // 0..7  (gather lane within row)

    // ---- stage f16 weights, transposed [N'][K] with gate-interleaved column permutation:
    //      c' = wave*64 + gate*16 + unitInWave  ->  n_orig = gate*64 + (wave*16 + unitInWave)
    for (int idx = tid; idx < NGATE * K0TOT; idx += NTHREADS) {
        const int cp = idx / K0TOT, k = idx % K0TOT;
        const int g = (cp >> 4) & 3;
        const int u = ((cp >> 6) << 4) + (cp & 15);
        const int norig = g * 64 + u;
        float v;
        if (k < 128) v = (k < EMBD) ? W0[k * NGATE + norig] : 0.0f;
        else         v = U0[(k - 128) * NGATE + norig];
        wt0[cp * K0TOT + k] = (_Float16)v;
    }
    for (int idx = tid; idx < NGATE * K1TOT; idx += NTHREADS) {
        const int cp = idx / K1TOT, k = idx % K1TOT;
        const int g = (cp >> 4) & 3;
        const int u = ((cp >> 6) << 4) + (cp & 15);
        const int norig = g * 64 + u;
        const float v = (k < UNITS) ? W1[k * NGATE + norig]
                                    : U1[(k - UNITS) * NGATE + norig];
        wt1[cp * K1TOT + k] = (_Float16)v;
    }
    // zero A-buffer (x pad + both h parities)
    for (int idx = tid; idx < BM * LDA; idx += NTHREADS) ab[idx] = (_Float16)0.0f;

    // per-lane gate biases and register cell states
    const float4 bb0 = {b0[j], b0[64 + j], b0[128 + j], b0[192 + j]};
    const float4 bb1 = {b1[j], b1[64 + j], b1[128 + j], b1[192 + j]};
    v8f c0r = {}, c1r = {};

    // ---- initial gather x(0) ----
    {
        const int token = inputs[(bbase + mrow) * SEQ + 0];
        const float* erow = emb + (size_t)token * EMBD;
        for (int e = ecol; e < EMBD; e += 8)
            ab[mrow * LDA + e] = (_Float16)erow[e];
    }
    __syncthreads();

#pragma unroll 1
    for (int t = 0; t < SEQ; ++t) {
        const int pr = (t & 1) * 128;   // read-parity offset; write regions at (256-pr)/(320-pr)

        // ---- layer0: z0 = [x || h0_old] @ [W0;U0], gates in-register, h0_new -> opposite parity
        {
            const int abase[6] = {0, 32, 64, 96, 128 + pr, 160 + pr};
            v8f acc[4] = {{}, {}, {}, {}};
            gemm_gates<6>(ab, abase, wt0, K0TOT, n0, lane, acc);
            cell_update(acc, c0r, bb0, ab + mb * LDA + (256 - pr) + j);
        }
        __syncthreads();   // h0_new visible to all waves

        // ---- issue next-step embedding gather (consumed after gemm1 -> latency hidden) ----
        float xr[13];
        const bool havex = (t + 1 < SEQ);
        if (havex) {
            const int token = inputs[(bbase + mrow) * SEQ + t + 1];
            const float* erow = emb + (size_t)token * EMBD;
#pragma unroll
            for (int i = 0; i < 13; ++i) {
                const int e = ecol + i * 8;
                xr[i] = (e < EMBD) ? erow[e] : 0.0f;
            }
            if ((t + 2 < SEQ) && (ecol < 4)) {   // prefetch t+2 rows (4 cachelines cover 400B)
                const int tk2 = inputs[(bbase + mrow) * SEQ + t + 2];
                __builtin_prefetch(emb + (size_t)tk2 * EMBD + ecol * 32, 0, 3);
            }
        }

        // ---- layer1: z1 = [h0_new || h1_old] @ [W1;U1] ----
        {
            const int abase[4] = {256 - pr, 288 - pr, 192 + pr, 224 + pr};
            v8f acc[4] = {{}, {}, {}, {}};
            gemm_gates<4>(ab, abase, wt1, K1TOT, n0, lane, acc);
            cell_update(acc, c1r, bb1, ab + mb * LDA + (320 - pr) + j);
        }

        // ---- commit next-step x into the X region (gemm0 readers already past mid-barrier) ----
        if (havex) {
#pragma unroll
            for (int i = 0; i < 13; ++i) {
                const int e = ecol + i * 8;
                if (e < EMBD) ab[mrow * LDA + e] = (_Float16)xr[i];
            }
        }
        __syncthreads();   // x(t+1) + h1_new visible
    }

    // ---- head: out[b] = sigmoid(h1_final . Wout + bout) ----
    constexpr int H1FIN = ((SEQ - 1) & 1) ? 192 : 320;   // SEQ=80 -> 192
    if (tid < BM) {
        float s = bout[0];
#pragma unroll 8
        for (int jj = 0; jj < UNITS; ++jj)
            s += (float)ab[tid * LDA + H1FIN + jj] * Wout[jj];
        out[bbase + tid] = sigmoidf_(s);
    }
}

extern "C" void kernel_launch(void* const* d_in, const int* in_sizes, int n_in,
                              void* d_out, int out_size, void* d_ws, size_t ws_size,
                              hipStream_t stream) {
    const int*   inputs = (const int*)  d_in[0];
    const float* emb    = (const float*)d_in[1];
    const float* W0     = (const float*)d_in[2];
    const float* U0     = (const float*)d_in[3];
    const float* b0     = (const float*)d_in[4];
    const float* W1     = (const float*)d_in[5];
    const float* U1     = (const float*)d_in[6];
    const float* b1     = (const float*)d_in[7];
    const float* Wout   = (const float*)d_in[8];
    const float* bout   = (const float*)d_in[9];
    float* out = (float*)d_out;

    (void)in_sizes; (void)n_in; (void)out_size; (void)d_ws; (void)ws_size;

    hipFuncSetAttribute((const void*)lstm2_fused,
                        hipFuncAttributeMaxDynamicSharedMemorySize,
                        (int)SMEM_BYTES);

    lstm2_fused<<<BATCH / BM, NTHREADS, SMEM_BYTES, stream>>>(
        inputs, emb, W0, U0, b0, W1, U1, b1, Wout, bout, out);
}
